// ObjectClassifier_51539607552708
// MI455X (gfx1250) — compile-verified, matrix operand order checked by reading, not run on hardware
//
#include <hip/hip_runtime.h>

#define N_BOX 8192
#define MAX_KEEP 64
#define IOU_TH 0.01f
#define BN_EPS 1e-5f
#define FEAT 2048
#define EMB 200
#define POSF 128
#define K1 2376      // 2048 + 200 + 128
#define K1P 2400     // padded to multiple of 32
#define HID 1024
#define N4 150
#define N4P 160

typedef __attribute__((ext_vector_type(16))) __bf16 v16bf;
typedef __attribute__((ext_vector_type(8)))  float  v8f;
typedef __attribute__((ext_vector_type(4)))  unsigned int u32x4;

// ---------------------------------------------------------------------------
// 1) Descending argsort of scores via bitonic sort on packed 64-bit keys.
//    key = (~bits(score) << 32) | idx  -> ascending sort == stable desc argsort
// ---------------------------------------------------------------------------
__global__ void sort_kernel(const float* __restrict__ scores,
                            int* __restrict__ sortedIdx)
{
    __shared__ unsigned long long keys[N_BOX];   // 64 KB LDS
    const int tid = threadIdx.x, NT = blockDim.x;
    for (int t = tid; t < N_BOX; t += NT) {
        unsigned sb = __float_as_uint(scores[t]);      // scores in [0,1): monotonic bits
        keys[t] = ((unsigned long long)(~sb) << 32) | (unsigned)t;
    }
    __syncthreads();
    for (int size = 2; size <= N_BOX; size <<= 1) {
        for (int stride = size >> 1; stride > 0; stride >>= 1) {
            for (int t = tid; t < N_BOX / 2; t += NT) {
                int lo = ((t & ~(stride - 1)) << 1) | (t & (stride - 1));
                int hi = lo + stride;
                bool desc = (lo & size) != 0;
                unsigned long long a = keys[lo], b = keys[hi];
                if ((a > b) != desc) { keys[lo] = b; keys[hi] = a; }
            }
            __syncthreads();
        }
    }
    for (int t = tid; t < N_BOX; t += NT)
        sortedIdx[t] = (int)(keys[t] & 0xffffffffu);
}

// ---------------------------------------------------------------------------
// 2) Greedy NMS on sorted order; early-exit once 64 boxes survive (sorted by
//    score, so first 64 kept == top_k(masked, 64)). Pads with suppressed
//    entries in sorted order if fewer than 64 survive (matches top_k ties).
// ---------------------------------------------------------------------------
__global__ void nms_kernel(const float* __restrict__ boxes,
                           const int* __restrict__ sortedIdx,
                           int* __restrict__ sel)
{
    __shared__ unsigned char keep[N_BOX];
    __shared__ int s_cnt;
    __shared__ float bi[4];
    const int tid = threadIdx.x, NT = blockDim.x;
    for (int t = tid; t < N_BOX; t += NT) keep[t] = 1;
    if (tid == 0) s_cnt = 0;
    __syncthreads();
    for (int i = 0; i < N_BOX; ++i) {
        if (!keep[i]) continue;                      // uniform (LDS) -> no divergence
        if (tid == 0) {
            int oi = sortedIdx[i];
            sel[s_cnt++] = oi;
            bi[0] = boxes[oi * 4 + 0]; bi[1] = boxes[oi * 4 + 1];
            bi[2] = boxes[oi * 4 + 2]; bi[3] = boxes[oi * 4 + 3];
        }
        __syncthreads();
        const float x1 = bi[0], y1 = bi[1], x2 = bi[2], y2 = bi[3];
        const float areai = (x2 - x1) * (y2 - y1);
        for (int j = i + 1 + tid; j < N_BOX; j += NT) {
            if (!keep[j]) continue;
            int oj = sortedIdx[j];
            float bx1 = boxes[oj * 4 + 0], by1 = boxes[oj * 4 + 1];
            float bx2 = boxes[oj * 4 + 2], by2 = boxes[oj * 4 + 3];
            float areaj = (bx2 - bx1) * (by2 - by1);
            float ix = fminf(x2, bx2) - fmaxf(x1, bx1);
            float iy = fminf(y2, by2) - fmaxf(y1, by1);
            float inter = fmaxf(ix, 0.f) * fmaxf(iy, 0.f);
            float iou = inter / (areai + areaj - inter + 1e-9f);
            if (iou > IOU_TH) keep[j] = 0;
        }
        __syncthreads();
        if (s_cnt >= MAX_KEEP) break;
    }
    __syncthreads();
    if (tid == 0) {
        int c = s_cnt;
        for (int i = 0; i < N_BOX && c < MAX_KEEP; ++i)
            if (!keep[i]) sel[c++] = sortedIdx[i];
    }
}

// ---------------------------------------------------------------------------
// 3) Weight conversion: f32 W[K][N] -> bf16 W^T[Npad][Kpad], zero padded.
//    Gives contiguous K-runs per output column -> 16B fragment loads in GEMM.
// ---------------------------------------------------------------------------
__global__ void convert_wt_kernel(const float* __restrict__ W,
                                  __bf16* __restrict__ WT,
                                  int K, int N, int Kp, int Np)
{
    long i = (long)blockIdx.x * blockDim.x + threadIdx.x;
    if (i >= (long)Np * Kp) return;
    int n = (int)(i / Kp), k = (int)(i - (long)n * Kp);
    float v = (k < K && n < N) ? W[(size_t)k * N + n] : 0.f;
    WT[i] = (__bf16)v;
}

// ---------------------------------------------------------------------------
// 4) Build X (64 x K1P bf16): [features | embed | relu(bn(center_size)@posW+b)]
//    plus labels/scores into the d_out tail.
// ---------------------------------------------------------------------------
__global__ void build_x_kernel(const float* __restrict__ boxes,
                               const float* __restrict__ scores,
                               const int* __restrict__ labels,
                               const float* __restrict__ features,
                               const float* __restrict__ embed_w,
                               const float* __restrict__ bn_pos,
                               const float* __restrict__ pos_w,
                               const float* __restrict__ pos_b,
                               const int* __restrict__ sel,
                               __bf16* __restrict__ X,
                               float* __restrict__ out)
{
    const int m = blockIdx.x, tid = threadIdx.x;
    const int s = sel[m];
    __shared__ float p[4];
    if (tid < 4) {
        float b0 = boxes[s * 4 + 0], b1 = boxes[s * 4 + 1];
        float b2 = boxes[s * 4 + 2], b3 = boxes[s * 4 + 3];
        float w = b2 - b0 + 1.f, hh = b3 - b1 + 1.f;
        float cx = b0 + 0.5f * w, cy = b1 + 0.5f * hh;
        float v = (tid == 0) ? cx : (tid == 1) ? cy : (tid == 2) ? w : hh;
        float g = bn_pos[tid], bt = bn_pos[4 + tid];
        float mn = bn_pos[8 + tid], vr = bn_pos[12 + tid];
        p[tid] = (v - mn) * rsqrtf(vr + BN_EPS) * g + bt;
    }
    __syncthreads();
    __bf16* xr = X + (size_t)m * K1P;
    for (int k = tid; k < FEAT; k += blockDim.x)
        xr[k] = (__bf16)features[(size_t)s * FEAT + k];
    const int lab = labels[s];
    for (int k = tid; k < EMB; k += blockDim.x)
        xr[FEAT + k] = (__bf16)embed_w[(size_t)(lab - 1) * EMB + k];
    for (int k = tid; k < POSF; k += blockDim.x) {
        float acc = pos_b[k];
#pragma unroll
        for (int j = 0; j < 4; ++j) acc += p[j] * pos_w[j * POSF + k];
        xr[FEAT + EMB + k] = (__bf16)fmaxf(acc, 0.f);
    }
    for (int k = K1 + tid; k < K1P; k += blockDim.x) xr[k] = (__bf16)0.f;
    if (tid == 0) {
        out[64 * N4 + m]      = (float)lab;   // obj_labels
        out[64 * N4 + 64 + m] = scores[s];    // obj_scores
    }
}

// ---------------------------------------------------------------------------
// 5) bf16 WMMA GEMM + fused bias/BN/ReLU. One wave per 16x16 output tile.
//    A: 64 x Kp row-major bf16.  WT: Npad x Kp row-major bf16 (pre-transposed).
//    16-bit A 16x32 fragment: elems 0..7 = K h*8.., elems 8..15 = K 16+h*8..
//    (B mirrors with N<->M).  C/D: elem e, lane -> M = 8*h + e, N = lane&15.
// ---------------------------------------------------------------------------
__global__ void gemm_wmma_kernel(const __bf16* __restrict__ A,
                                 const __bf16* __restrict__ WT,
                                 const float* __restrict__ bias,
                                 const float* __restrict__ bn,   // [4][Nout] or null
                                 __bf16* __restrict__ outB,
                                 float* __restrict__ outF,
                                 int Kp, int Nout, int out_stride)
{
    const int lane = threadIdx.x;
    const int h = lane >> 4, r = lane & 15;
    const int tn = blockIdx.x * 16, tm = blockIdx.y * 16;
    const __bf16* __restrict__ arow = A  + (size_t)(tm + r) * Kp + h * 8;
    const __bf16* __restrict__ brow = WT + (size_t)(tn + r) * Kp + h * 8;

    v8f acc = {};
    for (int k0 = 0; k0 < Kp; k0 += 32) {
        union { u32x4 u[2]; v16bf v; } a, b;
        a.u[0] = *(const u32x4*)(arow + k0);
        a.u[1] = *(const u32x4*)(arow + k0 + 16);
        b.u[0] = *(const u32x4*)(brow + k0);
        b.u[1] = *(const u32x4*)(brow + k0 + 16);
        acc = __builtin_amdgcn_wmma_f32_16x16x32_bf16(
                  /*neg_a=*/false, a.v, /*neg_b=*/false, b.v,
                  /*c_mod=*/(short)0, acc, /*reuse_a=*/false, /*reuse_b=*/false);
    }

    const int n = tn + r;
    if (n >= Nout) return;
    const float bia = bias[n];
    float g = 1.f, bt = 0.f, mn = 0.f, inv = 1.f;
    if (bn) {
        g  = bn[n];            bt  = bn[Nout + n];
        mn = bn[2 * Nout + n]; inv = rsqrtf(bn[3 * Nout + n] + BN_EPS);
    }
#pragma unroll
    for (int e = 0; e < 8; ++e) {
        const int m = tm + h * 8 + e;
        float x = acc[e] + bia;
        if (bn) x = fmaxf((x - mn) * inv * g + bt, 0.f);
        if (outF) outF[(size_t)m * out_stride + n] = x;
        else      outB[(size_t)m * out_stride + n] = (__bf16)x;
    }
}

// ---------------------------------------------------------------------------
extern "C" void kernel_launch(void* const* d_in, const int* in_sizes, int n_in,
                              void* d_out, int out_size, void* d_ws, size_t ws_size,
                              hipStream_t stream)
{
    const float* boxes    = (const float*)d_in[0];
    const float* scores   = (const float*)d_in[1];
    const int*   labels   = (const int*)  d_in[2];
    const float* features = (const float*)d_in[3];
    const float* embed_w  = (const float*)d_in[4];
    const float* bn_pos   = (const float*)d_in[5];
    const float* pos_w    = (const float*)d_in[6];
    const float* pos_b    = (const float*)d_in[7];
    const float* lin1_w   = (const float*)d_in[8];
    const float* lin1_b   = (const float*)d_in[9];
    const float* bn1      = (const float*)d_in[10];
    const float* lin2_w   = (const float*)d_in[11];
    const float* lin2_b   = (const float*)d_in[12];
    const float* bn2      = (const float*)d_in[13];
    const float* lin3_w   = (const float*)d_in[14];
    const float* lin3_b   = (const float*)d_in[15];
    const float* bn3      = (const float*)d_in[16];
    const float* lin4_w   = (const float*)d_in[17];
    const float* lin4_b   = (const float*)d_in[18];
    float* out = (float*)d_out;

    char* ws = (char*)d_ws;
    size_t off = 0;
    auto take = [&](size_t bytes) {
        char* p = ws + off;
        off = (off + bytes + 255) & ~(size_t)255;
        return p;
    };
    int*    sortedIdx = (int*)   take((size_t)N_BOX * 4);
    int*    sel       = (int*)   take((size_t)64 * 4);
    __bf16* X   = (__bf16*)take((size_t)64  * K1P * 2);
    __bf16* W1T = (__bf16*)take((size_t)HID * K1P * 2);
    __bf16* W2T = (__bf16*)take((size_t)HID * HID * 2);
    __bf16* W3T = (__bf16*)take((size_t)HID * HID * 2);
    __bf16* W4T = (__bf16*)take((size_t)N4P * HID * 2);
    __bf16* A1  = (__bf16*)take((size_t)64 * HID * 2);
    __bf16* A2  = (__bf16*)take((size_t)64 * HID * 2);
    __bf16* A3  = (__bf16*)take((size_t)64 * HID * 2);

    sort_kernel<<<1, 1024, 0, stream>>>(scores, sortedIdx);
    nms_kernel <<<1, 1024, 0, stream>>>(boxes, sortedIdx, sel);

    {
        long n;
        n = (long)HID * K1P;
        convert_wt_kernel<<<(n + 255) / 256, 256, 0, stream>>>(lin1_w, W1T, K1, HID, K1P, HID);
        n = (long)HID * HID;
        convert_wt_kernel<<<(n + 255) / 256, 256, 0, stream>>>(lin2_w, W2T, HID, HID, HID, HID);
        convert_wt_kernel<<<(n + 255) / 256, 256, 0, stream>>>(lin3_w, W3T, HID, HID, HID, HID);
        n = (long)N4P * HID;
        convert_wt_kernel<<<(n + 255) / 256, 256, 0, stream>>>(lin4_w, W4T, HID, N4, HID, N4P);
    }

    build_x_kernel<<<64, 256, 0, stream>>>(boxes, scores, labels, features, embed_w,
                                           bn_pos, pos_w, pos_b, sel, X, out);

    gemm_wmma_kernel<<<dim3(HID / 16, 4), 32, 0, stream>>>(X,  W1T, lin1_b, bn1,
                                                           A1, nullptr, K1P, HID, HID);
    gemm_wmma_kernel<<<dim3(HID / 16, 4), 32, 0, stream>>>(A1, W2T, lin2_b, bn2,
                                                           A2, nullptr, HID, HID, HID);
    gemm_wmma_kernel<<<dim3(HID / 16, 4), 32, 0, stream>>>(A2, W3T, lin3_b, bn3,
                                                           A3, nullptr, HID, HID, HID);
    gemm_wmma_kernel<<<dim3(N4P / 16, 4), 32, 0, stream>>>(A3, W4T, lin4_b, nullptr,
                                                           nullptr, out, HID, N4, N4);
}
